// InfoNceLoss_1236950581990
// MI455X (gfx1250) — matrix operand order repaired and checked by reading, not compiled, not run on hardware
//
#include <hip/hip_runtime.h>
#include <hip/hip_bf16.h>
#include <math.h>

// Problem constants from the reference: p1,p2 = [4096,256] f32, TAU = 0.5
#define N_ROWS 4096
#define TWO_N  8192
#define DIM    256
#define INV_TAU 2.0f
#define NGROUPS 128           // 8192 cols / 64 cols-per-block
#define MTILES  512           // 8192 rows / 16 rows-per-wave
#define SLOTS   512           // 16-column-granularity partial slots per row
#define NT      4             // 16x16 C tiles per wave along N
#define KSTEPS  (DIM / 32)    // 8
#define BSTRIDE 264           // LDS halves per B column (256 + 8 pad: bank-conflict-free)

typedef _Float16 v16h __attribute__((ext_vector_type(16)));
typedef _Float16 h8   __attribute__((ext_vector_type(8)));
typedef float    v8f  __attribute__((ext_vector_type(8)));

static __device__ __forceinline__ v16h load_frag(const _Float16* p0,
                                                 const _Float16* p1) {
    v16h r;
    *(h8*)&r       = *(const h8*)p0;
    *((h8*)&r + 1) = *(const h8*)p1;
    return r;
}

// ---------------------------------------------------------------------------
// Kernel 1: L2-normalize rows of [p1; p2] -> f16 z. One wave per row.
// ---------------------------------------------------------------------------
__global__ __launch_bounds__(256) void nrm_kernel(const float* __restrict__ p1,
                                                  const float* __restrict__ p2,
                                                  _Float16* __restrict__ zh) {
    const int wid  = (blockIdx.x * blockDim.x + threadIdx.x) >> 5;
    const int lane = threadIdx.x & 31;
    if (wid >= TWO_N) return;
    const float* src = (wid < N_ROWS) ? (p1 + (size_t)wid * DIM)
                                      : (p2 + (size_t)(wid - N_ROWS) * DIM);
    float v[8];
    float ss = 0.0f;
#pragma unroll
    for (int j = 0; j < 8; ++j) { v[j] = src[j * 32 + lane]; ss += v[j] * v[j]; }
#pragma unroll
    for (int off = 16; off >= 1; off >>= 1) ss += __shfl_xor(ss, off);
    const float inv = 1.0f / fmaxf(sqrtf(ss), 1e-12f);
    _Float16* dst = zh + (size_t)wid * DIM;
#pragma unroll
    for (int j = 0; j < 8; ++j) dst[j * 32 + lane] = (_Float16)(v[j] * inv);
}

// ---------------------------------------------------------------------------
// Kernel 2: upper-triangle-only sim GEMM (WMMA f16->f32), B staged in LDS.
//   * Block = 8 waves x 16 rows sharing one 64-col B tile (staged once into
//     LDS, padded stride -> conflict-free ds_load_b128 fragment reads).
//   * K-loop: 2 prefetched global A loads + 8 LDS B loads + 4 WMMAs per step.
//   * Symmetric crediting: sub-tile (tm,tn): tn>tm -> rows (slot tn) AND
//     transposed cols (slot tm); tn==tm -> rows only with i==j mask; tn<tm
//     -> discarded. Each partial[i][s] has exactly one writer (deterministic).
// ---------------------------------------------------------------------------
__global__ __launch_bounds__(256) void simgemm_kernel(const _Float16* __restrict__ zh,
                                                      float* __restrict__ partial) {
    const int ng  = blockIdx.x & (NGROUPS - 1);
    const int bm  = blockIdx.x >> 7;              // m-tile group (8 tiles)
    const int tn0 = ng * 4;                       // first 16-col tile of group
    // whole block below the diagonal? (uniform in blockIdx -> barrier-safe)
    if (tn0 + NT - 1 < bm * 8) return;

    const int lane = threadIdx.x & 31;
    const int w    = threadIdx.x >> 5;            // wave in block: 0..7
    const int tm   = bm * 8 + w;                  // per-wave 16-row tile
    const int nb   = ng * 64;
    const int mb   = tm * 16;
    const int l15  = lane & 15;
    const int lhi  = lane >> 4;

    // ---- stage the shared 64-col x 256-K B tile into LDS (all waves) ----
    __shared__ _Float16 Bs[64 * BSTRIDE];         // 33792 B
    {
        const int col  = threadIdx.x >> 2;        // 0..63
        const int ksec = (threadIdx.x & 3) * 64;  // 0,64,128,192
        const _Float16* gsrc = zh + (size_t)(nb + col) * DIM + ksec;
        _Float16* ldst = &Bs[col * BSTRIDE + ksec];
#pragma unroll
        for (int j = 0; j < 8; ++j)
            *(h8*)(ldst + j * 8) = *(const h8*)(gsrc + j * 8);
    }
    __syncthreads();

    // per-wave exit AFTER the barrier
    if (tn0 + NT - 1 < tm) return;

    // A fragment (16x32 f16, ISA 7.12.2): lane row = mb + (lane&15);
    // lanes 0-15 hold K {0..7,16..23}, lanes 16-31 hold K {8..15,24..31}.
    const _Float16* arow = zh + (size_t)(mb + l15) * DIM + lhi * 8;

    // B fragment from LDS (32x16): col = 16t + (lane&15); lanes 0-15 hold
    // K {0..7,16..23}, lanes 16-31 hold K {8..15,24..31} (matches A).
    const _Float16* bl[NT];
#pragma unroll
    for (int t = 0; t < NT; ++t)
        bl[t] = &Bs[(16 * t + l15) * BSTRIDE + lhi * 8];

    v8f c[NT];
#pragma unroll
    for (int t = 0; t < NT; ++t)
#pragma unroll
        for (int e = 0; e < 8; ++e) c[t][e] = 0.0f;

    // ---- pipelined K loop: global A double-buffered, B from LDS ----
    v16h a_cur = load_frag(arow, arow + 16);
    v16h b_cur[NT];
#pragma unroll
    for (int t = 0; t < NT; ++t)
        b_cur[t] = load_frag(bl[t], bl[t] + 16);

#pragma unroll
    for (int ks = 0; ks < KSTEPS; ++ks) {
        v16h a_nxt;
        v16h b_nxt[NT];
        if (ks < KSTEPS - 1) {
            const int k0n = (ks + 1) * 32;
            a_nxt = load_frag(arow + k0n, arow + k0n + 16);
#pragma unroll
            for (int t = 0; t < NT; ++t)
                b_nxt[t] = load_frag(bl[t] + k0n, bl[t] + k0n + 16);
        }
#pragma unroll
        for (int t = 0; t < NT; ++t)
            c[t] = __builtin_amdgcn_wmma_f32_16x16x32_f16(
                /*neg_a=*/false, a_cur, /*neg_b=*/false, b_cur[t],
                /*c_mod=*/(short)0, c[t], /*reuse_a=*/false, /*reuse_b=*/false);
        if (ks < KSTEPS - 1) {
            a_cur = a_nxt;
#pragma unroll
            for (int t = 0; t < NT; ++t) b_cur[t] = b_nxt[t];
        }
    }

    // Epilogue. C layout: VGPR r -> lanes 0-15 hold M=mb+r, lanes 16-31 hold
    // M=mb+8+r; N = nb + 16t + (lane&15).
#pragma unroll
    for (int t = 0; t < NT; ++t) {
        const int tn = tn0 + t;
        if (tn < tm) continue;                   // lower triangle: discard
        const bool diag = (tn == tm);
        const int col = nb + 16 * t + l15;       // this lane's column
        float cacc = 0.0f;                       // column accumulator
#pragma unroll
        for (int r = 0; r < 8; ++r) {
            const int row = mb + r + 8 * lhi;
            float e = __expf(INV_TAU * c[t][r]);
            if (diag && row == col) e = 0.0f;    // mask i==j
            cacc += e;
            // row-credit: reduce the 16 columns across this half-wave
            float racc = e;
#pragma unroll
            for (int off = 1; off < 16; off <<= 1) racc += __shfl_xor(racc, off);
            if (l15 == 0) partial[(size_t)row * SLOTS + tn] = racc;
        }
        if (!diag) {
            // col-credit (transpose): combine both halves (all 16 rows)
            cacc += __shfl_xor(cacc, 16);
            if (lhi == 0) partial[(size_t)col * SLOTS + tm] = cacc;
        }
    }
}

// ---------------------------------------------------------------------------
// Kernel 3: parallel per-row finalize. One wave per row i:
//   denom_i = sum_s partial[i][s];  pos_i = z_i . z_{i +/- N}
//   rowval[i] = log(denom_i) - pos_i/tau
// ---------------------------------------------------------------------------
__global__ __launch_bounds__(256) void rowval_kernel(const _Float16* __restrict__ zh,
                                                     const float* __restrict__ partial,
                                                     float* __restrict__ rowval) {
    const int i    = (blockIdx.x * blockDim.x + threadIdx.x) >> 5;
    const int lane = threadIdx.x & 31;
    if (i >= TWO_N) return;

    const float* pp = partial + (size_t)i * SLOTS;
    float s = 0.0f;
#pragma unroll
    for (int k = 0; k < SLOTS / 32; ++k) s += pp[lane + k * 32];

    const int pair = (i < N_ROWS) ? (i + N_ROWS) : (i - N_ROWS);
    h8 xa = *(const h8*)(zh + (size_t)i * DIM + lane * 8);
    h8 xb = *(const h8*)(zh + (size_t)pair * DIM + lane * 8);
    float dot = 0.0f;
#pragma unroll
    for (int e = 0; e < 8; ++e) dot += (float)xa[e] * (float)xb[e];

#pragma unroll
    for (int off = 16; off >= 1; off >>= 1) {
        s   += __shfl_xor(s, off);
        dot += __shfl_xor(dot, off);
    }
    if (lane == 0) rowval[i] = __logf(s) - INV_TAU * dot;
}

// ---------------------------------------------------------------------------
// Kernel 4: tiny deterministic tree-reduce of 8192 rowvals -> scalar loss.
// ---------------------------------------------------------------------------
__global__ __launch_bounds__(1024) void reduce_kernel(const float* __restrict__ rowval,
                                                      float* __restrict__ out) {
    __shared__ float red[1024];
    const int tid = threadIdx.x;
    float local = 0.0f;
#pragma unroll
    for (int k = 0; k < TWO_N / 1024; ++k) local += rowval[tid + k * 1024];
    red[tid] = local;
    __syncthreads();
    for (int s = 512; s > 0; s >>= 1) {
        if (tid < s) red[tid] += red[tid + s];
        __syncthreads();
    }
    if (tid == 0) out[0] = red[0] / (float)TWO_N;
}

// ---------------------------------------------------------------------------
extern "C" void kernel_launch(void* const* d_in, const int* in_sizes, int n_in,
                              void* d_out, int out_size, void* d_ws, size_t ws_size,
                              hipStream_t stream) {
    (void)in_sizes; (void)n_in; (void)out_size; (void)ws_size;
    const float* p1 = (const float*)d_in[0];
    const float* p2 = (const float*)d_in[1];

    char* ws = (char*)d_ws;
    _Float16* zh   = (_Float16*)ws;                                  // 4 MB
    float* partial = (float*)(ws + (size_t)TWO_N * DIM * 2);         // 16 MB
    float* rowval  = (float*)(ws + (size_t)TWO_N * DIM * 2
                                 + (size_t)TWO_N * SLOTS * 4);       // 32 KB

    // 1) normalize: 8 waves/block, 1 row per wave -> 1024 blocks
    nrm_kernel<<<TWO_N / 8, 256, 0, stream>>>(p1, p2, zh);

    // 2) upper-triangle WMMA sim-GEMM (B via LDS) with symmetric crediting
    simgemm_kernel<<<(MTILES / 8) * NGROUPS, 256, 0, stream>>>(zh, partial);

    // 3) per-row denom + positive-pair dot (1 wave/row -> 1024 blocks)
    rowval_kernel<<<TWO_N / 8, 256, 0, stream>>>(zh, partial, rowval);

    // 4) scalar loss
    reduce_kernel<<<1, 1024, 0, stream>>>(rowval, (float*)d_out);
}
